// TransformerConv_Encoder_85950885527601
// MI455X (gfx1250) — matrix operand order repaired and checked by reading, not compiled, not run on hardware
//
#include <hip/hip_runtime.h>
#include <hip/hip_bf16.h>

// ---------------- problem constants ----------------
#define NNODES 10000
#define NEDGES 160000
#define CIN    128
#define COUT   64
#define NHEAD  5
#define HQO    (NHEAD * COUT)      // 320
#define FTOT   (3 * HQO + COUT)    // 1024 packed: [Q(320)|K(320)|V(320)|S(64)]
#define QOFF   0
#define KOFF   HQO
#define VOFF   (2 * HQO)
#define SOFF   (3 * HQO)

#define COLBLK 4                   // col tiles per wave  -> 64 features per wave
#define ROWWAVES 8                 // row tiles per block (one per wave)
#define LDSPAD 8                   // halfs of padding per B row (bank skew)

typedef __attribute__((ext_vector_type(16))) _Float16 v16h;
typedef __attribute__((ext_vector_type(8)))  _Float16 v8h;
typedef __attribute__((ext_vector_type(8)))  float    v8f;

// ---------------- helpers ----------------
__device__ __forceinline__ void atomicMaxF(float* addr, float v) {
    // monotonic int-mapping trick (valid for all finite floats incl. -inf init)
    if (v >= 0.0f) atomicMax((int*)addr, __float_as_int(v));
    else           atomicMin((unsigned int*)addr, __float_as_uint(v));
}

// ---------------- packing kernels ----------------
__global__ void pack_x_f16(const float* __restrict__ x, _Float16* __restrict__ xh, int n) {
    int i = blockIdx.x * blockDim.x + threadIdx.x;
    if (i < n) xh[i] = (_Float16)x[i];
}

// Wt[m][k] = W[k][m], f16, m in [0,1024)
__global__ void pack_w_f16(const float* __restrict__ qw, const float* __restrict__ kw,
                           const float* __restrict__ vw, const float* __restrict__ sw,
                           _Float16* __restrict__ Wt, int K) {
    int idx = blockIdx.x * blockDim.x + threadIdx.x;
    if (idx >= FTOT * K) return;
    int m = idx / K, k = idx - m * K;
    float w;
    if (m < HQO)          w = qw[(size_t)k * HQO + m];
    else if (m < 2 * HQO) w = kw[(size_t)k * HQO + (m - HQO)];
    else if (m < 3 * HQO) w = vw[(size_t)k * HQO + (m - 2 * HQO)];
    else                  w = sw[(size_t)k * COUT + (m - 3 * HQO)];
    Wt[idx] = (_Float16)w;
}

__global__ void pack_b_f32(const float* __restrict__ qb, const float* __restrict__ kb,
                           const float* __restrict__ vb, const float* __restrict__ sb,
                           float* __restrict__ bias) {
    int m = blockIdx.x * blockDim.x + threadIdx.x;
    if (m >= FTOT) return;
    float b;
    if (m < HQO)          b = qb[m];
    else if (m < 2 * HQO) b = kb[m - HQO];
    else if (m < 3 * HQO) b = vb[m - 2 * HQO];
    else                  b = sb[m - 3 * HQO];
    bias[m] = b;
}

// ---------------- WMMA GEMM: C[M x 1024] = A[M x K] * Bt^T + bias ----------------
// A row-major f16, Bt is [1024 x K] f16 (transposed weights).
// Block = 8 waves; block stages a 64-feature x K weight slab in LDS (padded rows),
// each wave computes a 16x64 output strip (4 WMMA accumulators sharing one A frag).
__global__ __launch_bounds__(256)
void gemm_wmma_qkvs(const _Float16* __restrict__ A, const _Float16* __restrict__ Bt,
                    const float* __restrict__ bias, float* __restrict__ C,
                    int M, int K) {
    __shared__ _Float16 Bs[64 * (CIN + LDSPAD)];        // max 17408 B
    const int tid   = threadIdx.x;
    const int cb    = blockIdx.x & 15;                  // 16 col blocks of 64 features
    const int rb    = blockIdx.x >> 4;                  // row block (8 row tiles)
    const int ldsK  = K + LDSPAD;

    // cooperative load: Bs[m][0..K) = Bt[cb*64 + m][0..K), 8 halfs per uint4
    {
        const uint4* bsrc = (const uint4*)(Bt + (size_t)cb * 64 * K);
        const int kv = K >> 3;                          // uint4 per row
        for (int i = tid; i < 64 * kv; i += 256) {
            int m = i / kv, c = i - m * kv;
            *((uint4*)(Bs + m * ldsK) + c) = bsrc[i];
        }
    }
    __syncthreads();

    int nt = rb * ROWWAVES + (tid >> 5);
    if (nt * 16 >= M) return;                           // whole-wave exit (after barrier)
    int lane = tid & 31;
    int hs = lane >> 4;                                 // half-wave select
    int lr = lane & 15;

    const _Float16* arow = A + (size_t)(nt * 16 + lr) * K;   // A row (M = lr)
    v8f cacc[COLBLK] = {};

    for (int k0 = 0; k0 < K; k0 += 32) {
        // A 16x32 f16 frag: lanes<16 hold K {0..7,16..23}; lanes>=16 hold {8..15,24..31}
        union { v16h v; v8h h8[2]; } a;
        a.h8[0] = *(const v8h*)(arow + k0 + 8 * hs);
        a.h8[1] = *(const v8h*)(arow + k0 + 16 + 8 * hs);
        #pragma unroll
        for (int t = 0; t < COLBLK; ++t) {
            // B 32x16 frag from LDS: lanes<16 hold K {0..7,16..23} of col lr; lanes>=16 K {8..15,24..31}
            union { v16h v; v8h h8[2]; } b;
            const _Float16* bb = Bs + (size_t)(t * 16 + lr) * ldsK + k0 + 8 * hs;
            b.h8[0] = *(const v8h*)bb;
            b.h8[1] = *(const v8h*)(bb + 16);
            cacc[t] = __builtin_amdgcn_wmma_f32_16x16x32_f16(false, a.v, false, b.v,
                                                             (short)0, cacc[t], false, false);
        }
    }

    // C/D f32 layout: VGPR j -> row (j + 8*hs), col lr
    const int fbase = cb * 64;
    #pragma unroll
    for (int t = 0; t < COLBLK; ++t) {
        float bv = bias[fbase + t * 16 + lr];
        float* crow = C + (size_t)(nt * 16 + 8 * hs) * FTOT + fbase + t * 16 + lr;
        #pragma unroll
        for (int j = 0; j < 8; ++j)
            crow[(size_t)j * FTOT] = cacc[t][j] + bv;
    }
}

// ---------------- attention kernels ----------------
__global__ void init_accum(float* __restrict__ nmax, float* __restrict__ den,
                           float* __restrict__ acc) {
    int i = blockIdx.x * blockDim.x + threadIdx.x;
    if (i < NNODES * NHEAD * COUT) acc[i] = 0.0f;
    if (i < NNODES * NHEAD) { den[i] = 0.0f; nmax[i] = -3.0e38f; }
}

__global__ void edge_logits(const float* __restrict__ QKVS, const int* __restrict__ ei,
                            float* __restrict__ alpha, float* __restrict__ nmax) {
    int idx = blockIdx.x * blockDim.x + threadIdx.x;
    if (idx >= NEDGES * NHEAD) return;
    int e = idx / NHEAD, h = idx - e * NHEAD;
    int s = ei[e], d = ei[NEDGES + e];
    const float4* q = (const float4*)(QKVS + (size_t)d * FTOT + QOFF + h * COUT);
    const float4* k = (const float4*)(QKVS + (size_t)s * FTOT + KOFF + h * COUT);
    float acc = 0.0f;
    #pragma unroll
    for (int i = 0; i < COUT / 4; ++i) {
        float4 qa = q[i], kb = k[i];
        acc += qa.x * kb.x + qa.y * kb.y + qa.z * kb.z + qa.w * kb.w;
    }
    float a = acc * 0.125f;                 // 1/sqrt(64)
    alpha[idx] = a;
    atomicMaxF(&nmax[d * NHEAD + h], a);
}

__global__ void edge_exp_scatter(const float* __restrict__ QKVS, const int* __restrict__ ei,
                                 const float* __restrict__ alpha, const float* __restrict__ nmax,
                                 float* __restrict__ den, float* __restrict__ acc) {
    int idx = blockIdx.x * blockDim.x + threadIdx.x;
    if (idx >= NEDGES * NHEAD) return;
    int e = idx / NHEAD, h = idx - e * NHEAD;
    int s = ei[e], d = ei[NEDGES + e];
    float ex = expf(alpha[idx] - nmax[d * NHEAD + h]);
    atomicAdd(&den[d * NHEAD + h], ex);
    const float4* v = (const float4*)(QKVS + (size_t)s * FTOT + VOFF + h * COUT);
    float* ap = acc + ((size_t)d * NHEAD + h) * COUT;
    #pragma unroll
    for (int i = 0; i < COUT / 4; ++i) {
        float4 vv = v[i];
        atomicAdd(&ap[4 * i + 0], ex * vv.x);
        atomicAdd(&ap[4 * i + 1], ex * vv.y);
        atomicAdd(&ap[4 * i + 2], ex * vv.z);
        atomicAdd(&ap[4 * i + 3], ex * vv.w);
    }
}

__global__ void node_finalize(const float* __restrict__ QKVS, const float* __restrict__ den,
                              const float* __restrict__ acc, float* __restrict__ out_f32,
                              _Float16* __restrict__ out_f16) {
    int idx = blockIdx.x * blockDim.x + threadIdx.x;
    if (idx >= NNODES * COUT) return;
    int n = idx / COUT, c = idx - n * COUT;
    float s = 0.0f;
    #pragma unroll
    for (int h = 0; h < NHEAD; ++h) {
        float dh = den[n * NHEAD + h];
        float av = acc[((size_t)n * NHEAD + h) * COUT + c];
        s += (dh > 0.0f) ? av / dh : 0.0f;          // empty segments contribute 0
    }
    float val = s * (1.0f / NHEAD) + QKVS[(size_t)n * FTOT + SOFF + c];
    val = fmaxf(val, 0.0f);                         // relu after both layers
    if (out_f32) out_f32[idx] = val;
    if (out_f16) out_f16[idx] = (_Float16)val;
}

// ---------------- launcher ----------------
extern "C" void kernel_launch(void* const* d_in, const int* in_sizes, int n_in,
                              void* d_out, int out_size, void* d_ws, size_t ws_size,
                              hipStream_t stream) {
    const float* x   = (const float*)d_in[0];
    const int*   ei  = (const int*)d_in[1];
    const float* q1w = (const float*)d_in[2],  *q1b = (const float*)d_in[3];
    const float* k1w = (const float*)d_in[4],  *k1b = (const float*)d_in[5];
    const float* v1w = (const float*)d_in[6],  *v1b = (const float*)d_in[7];
    const float* s1w = (const float*)d_in[8],  *s1b = (const float*)d_in[9];
    const float* q2w = (const float*)d_in[10], *q2b = (const float*)d_in[11];
    const float* k2w = (const float*)d_in[12], *k2b = (const float*)d_in[13];
    const float* v2w = (const float*)d_in[14], *v2b = (const float*)d_in[15];
    const float* s2w = (const float*)d_in[16], *s2b = (const float*)d_in[17];
    float* out = (float*)d_out;

    char* base = (char*)d_ws;
    size_t off = 0;
    auto alloc = [&](size_t bytes) -> void* {
        void* p = base + off;
        off = (off + bytes + 255) & ~(size_t)255;
        return p;
    };
    _Float16* xh    = (_Float16*)alloc((size_t)NNODES * CIN * 2);
    _Float16* Wt    = (_Float16*)alloc((size_t)FTOT * CIN * 2);       // fits both layers
    float*    bias  = (float*)   alloc((size_t)FTOT * 4);
    float*    QKVS  = (float*)   alloc((size_t)NNODES * FTOT * 4);
    float*    alpha = (float*)   alloc((size_t)NEDGES * NHEAD * 4);
    float*    nmax  = (float*)   alloc((size_t)NNODES * NHEAD * 4);
    float*    den   = (float*)   alloc((size_t)NNODES * NHEAD * 4);
    float*    acc   = (float*)   alloc((size_t)NNODES * NHEAD * COUT * 4);
    _Float16* hh    = (_Float16*)alloc((size_t)NNODES * COUT * 2);

    const int T = 256;
    const int rowblocks   = (NNODES / 16 + ROWWAVES - 1) / ROWWAVES;  // 79
    const int gemm_blocks = rowblocks * 16;                           // 1264
    const int eh_blocks   = (NEDGES * NHEAD + T - 1) / T;
    const int init_blocks = (NNODES * NHEAD * COUT + T - 1) / T;
    const int node_blocks = (NNODES * COUT + T - 1) / T;

    // ---------- layer 1 ----------
    pack_x_f16<<<(NNODES * CIN + T - 1) / T, T, 0, stream>>>(x, xh, NNODES * CIN);
    pack_w_f16<<<(FTOT * CIN + T - 1) / T, T, 0, stream>>>(q1w, k1w, v1w, s1w, Wt, CIN);
    pack_b_f32<<<(FTOT + T - 1) / T, T, 0, stream>>>(q1b, k1b, v1b, s1b, bias);
    gemm_wmma_qkvs<<<gemm_blocks, T, 0, stream>>>(xh, Wt, bias, QKVS, NNODES, CIN);
    init_accum<<<init_blocks, T, 0, stream>>>(nmax, den, acc);
    edge_logits<<<eh_blocks, T, 0, stream>>>(QKVS, ei, alpha, nmax);
    edge_exp_scatter<<<eh_blocks, T, 0, stream>>>(QKVS, ei, alpha, nmax, den, acc);
    node_finalize<<<node_blocks, T, 0, stream>>>(QKVS, den, acc, nullptr, hh);

    // ---------- layer 2 ----------
    pack_w_f16<<<(FTOT * COUT + T - 1) / T, T, 0, stream>>>(q2w, k2w, v2w, s2w, Wt, COUT);
    pack_b_f32<<<(FTOT + T - 1) / T, T, 0, stream>>>(q2b, k2b, v2b, s2b, bias);
    gemm_wmma_qkvs<<<gemm_blocks, T, 0, stream>>>(hh, Wt, bias, QKVS, NNODES, COUT);
    init_accum<<<init_blocks, T, 0, stream>>>(nmax, den, acc);
    edge_logits<<<eh_blocks, T, 0, stream>>>(QKVS, ei, alpha, nmax);
    edge_exp_scatter<<<eh_blocks, T, 0, stream>>>(QKVS, ei, alpha, nmax, den, acc);
    node_finalize<<<node_blocks, T, 0, stream>>>(QKVS, den, acc, out, nullptr);
}
